// MultiHeadCrossAttention_39144331935811
// MI455X (gfx1250) — compile-verified, matrix operand order checked by reading, not run on hardware
//
#include <hip/hip_runtime.h>
#include <hip/hip_bf16.h>

#define BDIM    16
#define CDIM    512
#define NDIM    1024   // H*W
#define SDIM    77
#define SPAD    96     // padded S (3 x 32 K-steps)
#define CTXD    768
#define NHEADS  8
#define HDIM    64
#define NGROUPS 8
#define EPSV    1e-5f

typedef __bf16 bf16;
typedef __attribute__((ext_vector_type(16))) __bf16 v16bf;
typedef __attribute__((ext_vector_type(8)))  __bf16 v8bf;
typedef __attribute__((ext_vector_type(8)))  float  v8f;

// CDNA5 16-bit fragment: lane-half h holds K = h*8..h*8+7 (elems 0..7) and
// K = 16+h*8..16+h*8+7 (elems 8..15)  ->  two contiguous 16-byte chunks.
static __device__ __forceinline__ v16bf frag_ld_bf16(const bf16* p, int half) {
    v8bf lo = *(const v8bf*)(p + half * 8);
    v8bf hi = *(const v8bf*)(p + 16 + half * 8);
    return __builtin_shufflevector(lo, hi, 0, 1, 2, 3, 4, 5, 6, 7,
                                           8, 9, 10, 11, 12, 13, 14, 15);
}

// ---------------------------------------------------------------------------
// 0a) fp32 -> bf16 bulk convert (weights), 8 elems/thread, b128 in/out
// ---------------------------------------------------------------------------
__global__ void mhca_cvt_kernel(const float* __restrict__ src,
                                bf16* __restrict__ dst, int n8) {
    int i = blockIdx.x * blockDim.x + threadIdx.x;
    if (i >= n8) return;
    v8f v = *(const v8f*)(src + (size_t)i * 8);
    v8bf o;
#pragma unroll
    for (int j = 0; j < 8; ++j) o[j] = (bf16)v[j];
    *(v8bf*)(dst + (size_t)i * 8) = o;
}

// ---------------------------------------------------------------------------
// 0b) context -> bf16 with zero padding to SPAD rows: ctx_bf[b][96][768]
// ---------------------------------------------------------------------------
__global__ void mhca_ctx_pad_kernel(const float* __restrict__ ctx,
                                    bf16* __restrict__ dst) {
    const int CPB = SPAD * CTXD / 8;          // chunks per batch
    int i = blockIdx.x * blockDim.x + threadIdx.x;
    if (i >= BDIM * CPB) return;
    int batch = i / CPB;
    int r     = i % CPB;
    int s     = r / (CTXD / 8);
    v8bf o;
    if (s < SDIM) {
        const float* p = ctx + ((size_t)batch * SDIM + s) * CTXD + (r % (CTXD / 8)) * 8;
        v8f v = *(const v8f*)p;
#pragma unroll
        for (int j = 0; j < 8; ++j) o[j] = (bf16)v[j];
    } else {
#pragma unroll
        for (int j = 0; j < 8; ++j) o[j] = (bf16)0.f;
    }
    *(v8bf*)(dst + (size_t)i * 8) = o;
}

// ---------------------------------------------------------------------------
// 1) GroupNorm; emit h transposed: h_t[b][n][c]  (bf16, rows contiguous in c)
// ---------------------------------------------------------------------------
__global__ void mhca_gn_kernel(const float* __restrict__ x,
                               const float* __restrict__ w,
                               const float* __restrict__ b,
                               bf16* __restrict__ h_t) {
    const int CPG = CDIM / NGROUPS;   // 64
    const int CNT = CPG * NDIM;       // 65536
    int batch = blockIdx.x / NGROUPS;
    int g     = blockIdx.x % NGROUPS;
    const float* xp = x + ((size_t)batch * CDIM + (size_t)g * CPG) * NDIM;
    __shared__ float red0[8], red1[8];

    float s0 = 0.f, s1 = 0.f;
    for (int i = threadIdx.x; i < CNT; i += blockDim.x) {   // contiguous reads
        float v = xp[i];
        s0 += v; s1 += v * v;
    }
    for (int off = 16; off > 0; off >>= 1) {
        s0 += __shfl_down(s0, off, 32);
        s1 += __shfl_down(s1, off, 32);
    }
    int lane = threadIdx.x & 31, wid = threadIdx.x >> 5;
    if (lane == 0) { red0[wid] = s0; red1[wid] = s1; }
    __syncthreads();
    if (threadIdx.x == 0) {
        float t0 = 0.f, t1 = 0.f;
        int nw = blockDim.x >> 5;
        for (int i = 0; i < nw; ++i) { t0 += red0[i]; t1 += red1[i]; }
        red0[0] = t0; red1[0] = t1;
    }
    __syncthreads();
    float mean = red0[0] / (float)CNT;
    float var  = red1[0] / (float)CNT - mean * mean;
    float rstd = rsqrtf(var + EPSV);

    bf16* hp = h_t + (size_t)batch * NDIM * CDIM;
    for (int i = threadIdx.x; i < CNT; i += blockDim.x) {   // contiguous writes
        int n  = i / CPG;
        int cl = i % CPG;
        int c  = g * CPG + cl;
        float v = (xp[(size_t)cl * NDIM + n] - mean) * rstd * w[c] + b[c];
        hp[(size_t)n * CDIM + c] = (bf16)v;
    }
}

// ---------------------------------------------------------------------------
// 2) KV GEMM (all-bf16): K -> kbuf[b][s][o] (o<512), V -> vt[b][o-512][s]
//    grid (SPAD/16, 2C/64, B), 4 waves, 16x16 tile per wave
// ---------------------------------------------------------------------------
__global__ void mhca_kv_gemm_kernel(const bf16* __restrict__ ctx_bf,
                                    const bf16* __restrict__ kvw_bf,
                                    const float* __restrict__ kvb,
                                    bf16* __restrict__ kbuf,
                                    bf16* __restrict__ vt) {
    int batch = blockIdx.z;
    int stile = blockIdx.x;
    int wid   = threadIdx.x >> 5;
    int lane  = threadIdx.x & 31;
    int half  = lane >> 4;
    int rc    = lane & 15;
    int sRow  = stile * 16 + rc;
    int oCol  = (blockIdx.y * 4 + wid) * 16 + rc;

    const bf16* Arow = ctx_bf + ((size_t)batch * SPAD + sRow) * CTXD; // zero-padded
    const bf16* Brow = kvw_bf + (size_t)oCol * CTXD;

    v8f acc = {};
    for (int k0 = 0; k0 < CTXD; k0 += 32) {
        v16bf a  = frag_ld_bf16(Arow + k0, half);
        v16bf bv = frag_ld_bf16(Brow + k0, half);   // B[k=c][n=o] = kv_w[o][c]
        acc = __builtin_amdgcn_wmma_f32_16x16x32_bf16(false, a, false, bv,
                                                      (short)0, acc, false, false);
    }
    float bias = kvb[oCol];
    if (oCol < CDIM) {                 // K half: [s][o] rows (uniform per block)
        bf16* kp = kbuf + (size_t)batch * SPAD * CDIM;
#pragma unroll
        for (int i = 0; i < 8; ++i) {
            int s = stile * 16 + half * 8 + i;
            float v = (s < SDIM) ? (acc[i] + bias) : 0.f;
            kp[(size_t)s * CDIM + oCol] = (bf16)v;
        }
    } else {                           // V half: transposed [d][s], 16B store
        bf16* vp = vt + ((size_t)batch * CDIM + (oCol - CDIM)) * SPAD;
        int s0 = stile * 16 + half * 8;
        v8bf ov;
#pragma unroll
        for (int i = 0; i < 8; ++i) {
            int s = s0 + i;
            ov[i] = (bf16)((s < SDIM) ? (acc[i] + bias) : 0.f);
        }
        *(v8bf*)(vp + s0) = ov;
    }
}

// ---------------------------------------------------------------------------
// 3) Q GEMM (bf16 weights, 16x32 per wave): q_t[b][n][o], 16B stores
//    grid (C/16, N/128, B), 4 waves
// ---------------------------------------------------------------------------
__global__ void mhca_q_gemm_kernel(const bf16* __restrict__ qw_bf,
                                   const float* __restrict__ qb,
                                   const bf16* __restrict__ h_t,
                                   bf16* __restrict__ q_t) {
    int batch = blockIdx.z;
    int otile = blockIdx.x;
    int wid   = threadIdx.x >> 5;
    int lane  = threadIdx.x & 31;
    int half  = lane >> 4;
    int rc    = lane & 15;
    int oRow  = otile * 16 + rc;
    int nCol0 = (blockIdx.y * 8 + wid * 2) * 16 + rc;
    int nCol1 = nCol0 + 16;
    const bf16* Arow = qw_bf + (size_t)oRow * CDIM;
    const bf16* B0   = h_t + ((size_t)batch * NDIM + nCol0) * CDIM; // [n][c]
    const bf16* B1   = h_t + ((size_t)batch * NDIM + nCol1) * CDIM;

    v8f acc0 = {}, acc1 = {};
    for (int k0 = 0; k0 < CDIM; k0 += 32) {
        __builtin_prefetch((const void*)(B0 + k0 + 64), 0, 1);
        v16bf a  = frag_ld_bf16(Arow + k0, half);   // shared across both tiles
        v16bf b0 = frag_ld_bf16(B0 + k0, half);
        v16bf b1 = frag_ld_bf16(B1 + k0, half);
        acc0 = __builtin_amdgcn_wmma_f32_16x16x32_bf16(false, a, false, b0,
                                                       (short)0, acc0, false, false);
        acc1 = __builtin_amdgcn_wmma_f32_16x16x32_bf16(false, a, false, b1,
                                                       (short)0, acc1, false, false);
    }
    const float scale = 0.125f; // 64^-0.5 folded into Q
    int o0 = otile * 16 + half * 8;
    v8bf ov0, ov1;
#pragma unroll
    for (int i = 0; i < 8; ++i) {
        float bias = qb[o0 + i];
        ov0[i] = (bf16)((acc0[i] + bias) * scale);
        ov1[i] = (bf16)((acc1[i] + bias) * scale);
    }
    *(v8bf*)(q_t + ((size_t)batch * NDIM + nCol0) * CDIM + o0) = ov0;
    *(v8bf*)(q_t + ((size_t)batch * NDIM + nCol1) * CDIM + o0) = ov1;
}

// ---------------------------------------------------------------------------
// 4) Attention per (b, head, 32 rows): QK^T -> masked softmax -> V^T x P^T
//    grid (N/32, HEADS, B), 4 waves
// ---------------------------------------------------------------------------
__global__ void mhca_attn_kernel(const bf16* __restrict__ q_t,
                                 const bf16* __restrict__ kbuf,
                                 const bf16* __restrict__ vt,
                                 bf16* __restrict__ o_t) {
    int batch = blockIdx.z, head = blockIdx.y, nblk = blockIdx.x;
    int wid  = threadIdx.x >> 5;
    int lane = threadIdx.x & 31;
    int half = lane >> 4;
    int rc   = lane & 15;

    __shared__ float              sL[32][SPAD];  // logits
    __shared__ __align__(16) bf16 sP[32][SPAD];  // probs, row-major [n][s]

    const bf16* Qp = q_t + ((size_t)batch * NDIM) * CDIM + head * HDIM;  // [n][d]
    const bf16* Kp = kbuf + (size_t)batch * SPAD * CDIM + head * HDIM;   // [s][d]
    const bf16* Vt = vt + ((size_t)batch * CDIM + head * HDIM) * SPAD;   // [d][s]

    // --- GEMM1: logits[32][96] = Q[32][64] x K^T ; 2x6 tiles, 3/wave ---
    for (int t = wid; t < 12; t += 4) {
        int mt = t / 6, st = t % 6;
        int nRow = nblk * 32 + mt * 16 + rc;
        int sCol = st * 16 + rc;
        const bf16* Arow = Qp + (size_t)nRow * CDIM;
        const bf16* Brow = Kp + (size_t)sCol * CDIM;  // B[k=d][n=s] = K[s][d]
        v8f acc = {};
#pragma unroll
        for (int k0 = 0; k0 < HDIM; k0 += 32) {
            v16bf a  = frag_ld_bf16(Arow + k0, half);
            v16bf bv = frag_ld_bf16(Brow + k0, half);
            acc = __builtin_amdgcn_wmma_f32_16x16x32_bf16(false, a, false, bv,
                                                          (short)0, acc, false, false);
        }
#pragma unroll
        for (int i = 0; i < 8; ++i)
            sL[mt * 16 + half * 8 + i][sCol] = acc[i];
    }
    __syncthreads();

    // --- masked softmax over s (one row per lane of wave 0) ---
    if (threadIdx.x < 32) {
        int m = threadIdx.x;
        float mx = -1e30f;
        for (int s = 0; s < SDIM; ++s) mx = fmaxf(mx, sL[m][s]);
        float sum = 0.f;
        for (int s = 0; s < SPAD; ++s) {
            float p = (s < SDIM) ? __expf(sL[m][s] - mx) : 0.f;
            sL[m][s] = p;
            sum += p;
        }
        float inv = 1.f / sum;
        for (int s = 0; s < SPAD; ++s) sP[m][s] = (bf16)(sL[m][s] * inv);
    }
    __syncthreads();

    // --- GEMM2 (swapped): out^T[d][n] = V^T[64][96] x P^T[96][32]
    //     D cols = n fixed per lane, rows = contiguous d -> 16B stores.
    for (int t = wid; t < 8; t += 4) {
        int dt = t >> 1, nt = t & 1;
        int dRow   = dt * 16 + rc;
        int nLocal = nt * 16 + rc;
        const bf16* Arow = Vt + (size_t)dRow * SPAD;
        const bf16* Brow = &sP[nLocal][0];
        v8f acc = {};
#pragma unroll
        for (int k0 = 0; k0 < SPAD; k0 += 32) {
            v16bf a  = frag_ld_bf16(Arow + k0, half);
            v16bf bv = frag_ld_bf16(Brow + k0, half);
            acc = __builtin_amdgcn_wmma_f32_16x16x32_bf16(false, a, false, bv,
                                                          (short)0, acc, false, false);
        }
        int n  = nblk * 32 + nLocal;
        int d0 = dt * 16 + half * 8;
        v8bf ov;
#pragma unroll
        for (int i = 0; i < 8; ++i) ov[i] = (bf16)acc[i];
        *(v8bf*)(o_t + ((size_t)batch * NDIM + n) * CDIM + head * HDIM + d0) = ov;
    }
}

// ---------------------------------------------------------------------------
// 5) Proj GEMM (bf16 weights, 16x32 per wave) + bias + residual, fp32 out
//    grid (C/16, N/128, B), 4 waves
// ---------------------------------------------------------------------------
__global__ void mhca_proj_kernel(const bf16* __restrict__ pw_bf,
                                 const float* __restrict__ pb,
                                 const bf16* __restrict__ o_t,
                                 const float* __restrict__ x,
                                 float* __restrict__ outp) {
    int batch = blockIdx.z;
    int otile = blockIdx.x;
    int wid   = threadIdx.x >> 5;
    int lane  = threadIdx.x & 31;
    int half  = lane >> 4;
    int rc    = lane & 15;
    int oRow  = otile * 16 + rc;
    int nCol0 = (blockIdx.y * 8 + wid * 2) * 16 + rc;
    int nCol1 = nCol0 + 16;
    const bf16* Arow = pw_bf + (size_t)oRow * CDIM;
    const bf16* B0   = o_t + ((size_t)batch * NDIM + nCol0) * CDIM; // [n][o]
    const bf16* B1   = o_t + ((size_t)batch * NDIM + nCol1) * CDIM;

    v8f acc0 = {}, acc1 = {};
    for (int k0 = 0; k0 < CDIM; k0 += 32) {
        __builtin_prefetch((const void*)(B0 + k0 + 64), 0, 1);
        v16bf a  = frag_ld_bf16(Arow + k0, half);
        v16bf b0 = frag_ld_bf16(B0 + k0, half);
        v16bf b1 = frag_ld_bf16(B1 + k0, half);
        acc0 = __builtin_amdgcn_wmma_f32_16x16x32_bf16(false, a, false, b0,
                                                       (short)0, acc0, false, false);
        acc1 = __builtin_amdgcn_wmma_f32_16x16x32_bf16(false, a, false, b1,
                                                       (short)0, acc1, false, false);
    }
    const float* xp = x + (size_t)batch * CDIM * NDIM;
    float* op = outp + (size_t)batch * CDIM * NDIM;
#pragma unroll
    for (int i = 0; i < 8; ++i) {
        int oo = otile * 16 + half * 8 + i;
        float bias = pb[oo];
        op[(size_t)oo * NDIM + nCol0] = acc0[i] + bias + xp[(size_t)oo * NDIM + nCol0];
        op[(size_t)oo * NDIM + nCol1] = acc1[i] + bias + xp[(size_t)oo * NDIM + nCol1];
    }
}

// ---------------------------------------------------------------------------
extern "C" void kernel_launch(void* const* d_in, const int* in_sizes, int n_in,
                              void* d_out, int out_size, void* d_ws, size_t ws_size,
                              hipStream_t stream) {
    (void)in_sizes; (void)n_in; (void)out_size; (void)ws_size;
    const float* x       = (const float*)d_in[0];
    const float* context = (const float*)d_in[1];
    const float* gn_w    = (const float*)d_in[2];
    const float* gn_b    = (const float*)d_in[3];
    const float* q_w     = (const float*)d_in[4];
    const float* q_b     = (const float*)d_in[5];
    const float* kv_w    = (const float*)d_in[6];
    const float* kv_b    = (const float*)d_in[7];
    const float* proj_w  = (const float*)d_in[8];
    const float* proj_b  = (const float*)d_in[9];
    float* out           = (float*)d_out;

    char* ws = (char*)d_ws;
    size_t off = 0;
    const size_t BCN = (size_t)BDIM * CDIM * NDIM;       // 8M elems
    const size_t KVN = (size_t)BDIM * SPAD * CDIM;       // 786K elems
    bf16* h_t    = (bf16*)(ws + off); off += BCN * 2;                    // 16 MB
    bf16* q_t    = (bf16*)(ws + off); off += BCN * 2;                    // 16 MB
    bf16* o_t    = (bf16*)(ws + off); off += BCN * 2;                    // 16 MB
    bf16* kbuf   = (bf16*)(ws + off); off += KVN * 2;                    // 1.5 MB
    bf16* vt     = (bf16*)(ws + off); off += KVN * 2;                    // 1.5 MB
    bf16* qw_bf  = (bf16*)(ws + off); off += (size_t)CDIM * CDIM * 2;    // 0.5 MB
    bf16* pw_bf  = (bf16*)(ws + off); off += (size_t)CDIM * CDIM * 2;    // 0.5 MB
    bf16* kvw_bf = (bf16*)(ws + off); off += (size_t)2 * CDIM * CTXD * 2;// 1.5 MB
    bf16* ctx_bf = (bf16*)(ws + off); off += (size_t)BDIM * SPAD * CTXD * 2; // 2.25 MB

    // one-shot bf16 conversions (weights + zero-padded context)
    int n8w = CDIM * CDIM / 8;
    mhca_cvt_kernel<<<(n8w + 255) / 256, 256, 0, stream>>>(q_w, qw_bf, n8w);
    mhca_cvt_kernel<<<(n8w + 255) / 256, 256, 0, stream>>>(proj_w, pw_bf, n8w);
    int n8kv = 2 * CDIM * CTXD / 8;
    mhca_cvt_kernel<<<(n8kv + 255) / 256, 256, 0, stream>>>(kv_w, kvw_bf, n8kv);
    int n8c = BDIM * SPAD * CTXD / 8;
    mhca_ctx_pad_kernel<<<(n8c + 255) / 256, 256, 0, stream>>>(context, ctx_bf);

    mhca_gn_kernel<<<BDIM * NGROUPS, 256, 0, stream>>>(x, gn_w, gn_b, h_t);
    mhca_kv_gemm_kernel<<<dim3(SPAD / 16, (2 * CDIM) / 64, BDIM), 128, 0, stream>>>(
        ctx_bf, kvw_bf, kv_b, kbuf, vt);
    mhca_q_gemm_kernel<<<dim3(CDIM / 16, NDIM / 128, BDIM), 128, 0, stream>>>(
        qw_bf, q_b, h_t, q_t);
    mhca_attn_kernel<<<dim3(NDIM / 32, NHEADS, BDIM), 128, 0, stream>>>(
        q_t, kbuf, vt, o_t);
    mhca_proj_kernel<<<dim3(CDIM / 16, NDIM / 128, BDIM), 128, 0, stream>>>(
        pw_bf, proj_b, o_t, x, out);
}
